// AnchorHead_49194555408620
// MI455X (gfx1250) — matrix operand order repaired and checked by reading, not compiled, not run on hardware
//
#include <hip/hip_runtime.h>

#define USE_ASYNC 1   // async global->LDS DMA pipeline (gfx1250 ASYNCcnt path)

typedef __bf16 bf16;
typedef __attribute__((ext_vector_type(16))) __bf16 v16bf;
typedef __attribute__((ext_vector_type(8)))  __bf16 v8bf;
typedef __attribute__((ext_vector_type(8)))  float   v8f;

#define CIN 256
#define BM 128
#define BN 64
#define LDA 72
#define LDB 72
#define ABUF (BM * LDA)   // elems per A stage buffer
#define BBUF (BN * LDB)   // elems per B stage buffer
#define NSTAGE 36         // 9 taps * (256/64) k-chunks

__global__ __launch_bounds__(256) void cvt_f32_bf16(const float* __restrict__ in,
                                                    bf16* __restrict__ out, int n) {
  int i = (blockIdx.x * 256 + threadIdx.x) * 4;
  if (i + 3 < n) {
    const float4 f = *(const float4*)(in + i);
    out[i + 0] = (bf16)f.x; out[i + 1] = (bf16)f.y;
    out[i + 2] = (bf16)f.z; out[i + 3] = (bf16)f.w;
  } else {
    for (int j = 0; j < 4; ++j) if (i + j < n) out[i + j] = (bf16)in[i + j];
  }
}

// w: [9][256][Cout] f32 (HWIO flattened)  ->  wt: [9][Cout][256] bf16
__global__ __launch_bounds__(256) void wtrans_bf16(const float* __restrict__ w,
                                                   bf16* __restrict__ wt, int Cout) {
  int i = blockIdx.x * 256 + threadIdx.x;
  int total = 9 * CIN * Cout;
  if (i < total) {
    int ci  = i & (CIN - 1);
    int t   = i >> 8;
    int co  = t % Cout;
    int tap = t / Cout;
    wt[i] = (bf16)w[(size_t)(tap * CIN + ci) * Cout + co];
  }
}

// Implicit-GEMM 3x3 SAME conv: 9 accumulated shifted 1x1 GEMMs, bf16 WMMA,
// double-buffered LDS with async global->LDS DMA overlapping WMMA compute.
template <bool RELU, bool OUTBF>
__global__ __launch_bounds__(256) void conv3x3_wmma(
    const bf16* __restrict__ in, const bf16* __restrict__ wt,
    const float* __restrict__ bias, bf16* __restrict__ outb,
    float* __restrict__ outf, int H, int W, int Cout, int pixels) {
  __shared__ __align__(16) bf16 sA[2 * ABUF];
  __shared__ __align__(16) bf16 sB[2 * BBUF];

  const int tid  = threadIdx.x;
  const int lane = tid & 31;
  const int wid  = tid >> 5;
  const int wm   = wid & 3;   // 4 waves along M (32 rows each)
  const int wn   = wid >> 2;  // 2 waves along N (32 cols each)
  const int mr   = lane & 15;
  const int kh   = lane >> 4;

  const int m_base = blockIdx.x * BM;
  const int n_base = blockIdx.y * BN;

  // ---- A loader: fixed per-thread pixel decode ----
  const int arow  = tid >> 1;
  const int ahalf = tid & 1;
  const int p     = m_base + arow;
  const bool pv   = p < pixels;
  const int pc    = pv ? p : 0;
  const int HW    = H * W;
  const int nImg  = pc / HW;
  const int rem   = pc - nImg * HW;
  const int hP    = rem / W;
  const int wP    = rem - hP * W;

  // ---- B loader ----
  const int brow    = tid >> 2;
  const int bq      = tid & 3;
  const int co_load = n_base + brow;
  const bool bvalid = co_load < Cout;

#if USE_ASYNC
  // LDS byte addresses (generic-pointer low 32 bits == LDS offset)
  const unsigned ldsA0 = (unsigned)(size_t)&sA[0] + (unsigned)(arow * LDA + (ahalf << 5)) * 2u;
  const unsigned ldsB0 = (unsigned)(size_t)&sB[0] + (unsigned)(brow * LDB + (bq << 4)) * 2u;
#endif

  // Issue data movement for one stage into LDS buffer (stage&1).
  auto issue = [&](int stage) {
    const int tap = stage >> 2;
    const int k0  = (stage & 3) << 6;
    const int buf = stage & 1;
    const int dy  = tap / 3 - 1;
    const int dx  = tap - (tap / 3) * 3 - 1;
    const int hh  = hP + dy;
    const int ww  = wP + dx;
    const bool av = pv && ((unsigned)hh < (unsigned)H) && ((unsigned)ww < (unsigned)W);

    if (bvalid && (stage & 3) == 0 && tap < 8)  // warm L2 for next tap's weights
      __builtin_prefetch(wt + ((size_t)((tap + 1) * Cout + co_load) << 8), 0, 1);

    // A tile: 128 rows x 64 bf16 (zero-filled at image borders)
    if (av) {
      const bf16* g = in + (((size_t)(nImg * H + hh) * W + ww) << 8) + (ahalf << 5) + k0;
#if USE_ASYNC
      const unsigned la = ldsA0 + (unsigned)buf * (ABUF * 2u);
      #pragma unroll
      for (int c2 = 0; c2 < 4; ++c2)
        asm volatile("global_load_async_to_lds_b128 %0, %1, off"
                     :: "v"(la + c2 * 16u), "v"((unsigned long long)(size_t)(g + c2 * 8))
                     : "memory");
#else
      #pragma unroll
      for (int c2 = 0; c2 < 4; ++c2)
        *(uint4*)&sA[buf * ABUF + arow * LDA + (ahalf << 5) + (c2 << 3)] =
            ((const uint4*)g)[c2];
#endif
    } else {
      const uint4 z = make_uint4(0u, 0u, 0u, 0u);
      #pragma unroll
      for (int c2 = 0; c2 < 4; ++c2)
        *(uint4*)&sA[buf * ABUF + arow * LDA + (ahalf << 5) + (c2 << 3)] = z;
    }

    // B tile: 64 rows (co) x 64 bf16 (ci), zero-filled past Cout
    if (bvalid) {
      const bf16* g = wt + ((size_t)(tap * Cout + co_load) << 8) + (bq << 4) + k0;
#if USE_ASYNC
      const unsigned lb = ldsB0 + (unsigned)buf * (BBUF * 2u);
      #pragma unroll
      for (int c2 = 0; c2 < 2; ++c2)
        asm volatile("global_load_async_to_lds_b128 %0, %1, off"
                     :: "v"(lb + c2 * 16u), "v"((unsigned long long)(size_t)(g + c2 * 8))
                     : "memory");
#else
      #pragma unroll
      for (int c2 = 0; c2 < 2; ++c2)
        *(uint4*)&sB[buf * BBUF + brow * LDB + (bq << 4) + (c2 << 3)] =
            ((const uint4*)g)[c2];
#endif
    } else {
      const uint4 z = make_uint4(0u, 0u, 0u, 0u);
      #pragma unroll
      for (int c2 = 0; c2 < 2; ++c2)
        *(uint4*)&sB[buf * BBUF + brow * LDB + (bq << 4) + (c2 << 3)] = z;
    }
  };

  v8f zero = {};
  v8f acc[2][2];
  acc[0][0] = zero; acc[0][1] = zero; acc[1][0] = zero; acc[1][1] = zero;

  // ---- pipeline prologue ----
  issue(0);
#if USE_ASYNC
  asm volatile("s_wait_asynccnt 0x0" ::: "memory");
#endif
  __syncthreads();

  for (int s = 0; s < NSTAGE; ++s) {
    if (s + 1 < NSTAGE) issue(s + 1);  // DMA next stage while computing this one

    const int ab = (s & 1) * ABUF;
    const int bb = (s & 1) * BBUF;
    #pragma unroll
    for (int ks = 0; ks < 64; ks += 32) {
      v16bf af[2], bfm[2];
      #pragma unroll
      for (int mi = 0; mi < 2; ++mi) {
        // A frag: lane<16 -> K 0..7 & 16..23 ; lane>=16 -> K 8..15 & 24..31
        const int row = (wm << 5) + (mi << 4) + mr;
        v8bf lo = *(const v8bf*)&sA[ab + row * LDA + ks + (kh << 3)];
        v8bf hi = *(const v8bf*)&sA[ab + row * LDA + ks + 16 + (kh << 3)];
        af[mi] = __builtin_shufflevector(lo, hi, 0, 1, 2, 3, 4, 5, 6, 7,
                                         8, 9, 10, 11, 12, 13, 14, 15);
      }
      #pragma unroll
      for (int ni = 0; ni < 2; ++ni) {
        // B frag: lane = column; lanes 0-15 hold K 0..7 & 16..23, lanes 16-31 K 8..15 & 24..31
        const int row = (wn << 5) + (ni << 4) + mr;
        v8bf lo = *(const v8bf*)&sB[bb + row * LDB + ks + (kh << 3)];
        v8bf hi = *(const v8bf*)&sB[bb + row * LDB + ks + 16 + (kh << 3)];
        bfm[ni] = __builtin_shufflevector(lo, hi, 0, 1, 2, 3, 4, 5, 6, 7,
                                          8, 9, 10, 11, 12, 13, 14, 15);
      }
      #pragma unroll
      for (int mi = 0; mi < 2; ++mi)
        #pragma unroll
        for (int ni = 0; ni < 2; ++ni)
          acc[mi][ni] = __builtin_amdgcn_wmma_f32_16x16x32_bf16(
              false, af[mi], false, bfm[ni], (short)0, acc[mi][ni], false, false);
    }

#if USE_ASYNC
    asm volatile("s_wait_asynccnt 0x0" ::: "memory");  // next-stage DMA landed
#endif
    __syncthreads();  // also waits DScnt (zero-fill stores + frag loads)
  }

  // ---- epilogue: bias (+ReLU), scatter per C/D layout (VGPR v -> M = v + 8*kh) ----
  const int prow0 = m_base + (wm << 5) + (kh << 3);
  const int col0  = n_base + (wn << 5) + mr;
  #pragma unroll
  for (int mi = 0; mi < 2; ++mi) {
    #pragma unroll
    for (int ni = 0; ni < 2; ++ni) {
      const int co = col0 + (ni << 4);
      if (co < Cout) {
        const float bb = bias[co];
        #pragma unroll
        for (int v = 0; v < 8; ++v) {
          const int pp = prow0 + (mi << 4) + v;
          if (pp < pixels) {
            float val = acc[mi][ni][v] + bb;
            if (RELU) val = fmaxf(val, 0.0f);
            const size_t o = (size_t)pp * Cout + co;
            if (OUTBF) outb[o] = (bf16)val;
            else       outf[o] = val;
          }
        }
      }
    }
  }
}

extern "C" void kernel_launch(void* const* d_in, const int* in_sizes, int n_in,
                              void* d_out, int out_size, void* d_ws, size_t ws_size,
                              hipStream_t stream) {
  const float* x[3]       = {(const float*)d_in[0], (const float*)d_in[1], (const float*)d_in[2]};
  const float* cls_w      = (const float*)d_in[3];
  const float* cls_b      = (const float*)d_in[4];
  const float* bbox_w     = (const float*)d_in[5];
  const float* bbox_b     = (const float*)d_in[6];
  const float* cls_out_w  = (const float*)d_in[7];
  const float* cls_out_b  = (const float*)d_in[8];
  const float* bbox_out_w = (const float*)d_in[9];
  const float* bbox_out_b = (const float*)d_in[10];
  float* out = (float*)d_out;

  // workspace layout
  const size_t BUFE = (size_t)8 * 128 * 128 * 256;   // max activation elems
  bf16* bufA = (bf16*)d_ws;
  bf16* bufB = bufA + BUFE;
  bf16* wreg = bufB + BUFE;
  const size_t WL = (size_t)9 * 256 * 256;           // elems per 256->256 layer
  bf16* cls_wt[4];  bf16* bbox_wt[4];
  for (int i = 0; i < 4; ++i) cls_wt[i]  = wreg + (size_t)i * WL;
  for (int i = 0; i < 4; ++i) bbox_wt[i] = wreg + (size_t)(4 + i) * WL;
  bf16* cls_out_wt  = wreg + 8 * WL;
  bf16* bbox_out_wt = cls_out_wt + (size_t)9 * 720 * 256;

  auto wlaunch = [&](const float* src, bf16* dst, int Cout) {
    int total = 9 * 256 * Cout;
    wtrans_bf16<<<(total + 255) / 256, 256, 0, stream>>>(src, dst, Cout);
  };
  for (int i = 0; i < 4; ++i) wlaunch(cls_w  + (size_t)i * WL, cls_wt[i],  256);
  for (int i = 0; i < 4; ++i) wlaunch(bbox_w + (size_t)i * WL, bbox_wt[i], 256);
  wlaunch(cls_out_w,  cls_out_wt,  720);
  wlaunch(bbox_out_w, bbox_out_wt, 36);

  const int Hs[3] = {128, 64, 32};
  size_t clsOff[3], boxOff[3], off = 0;
  for (int l = 0; l < 3; ++l) { clsOff[l] = off; off += (size_t)8 * Hs[l] * Hs[l] * 720; }
  for (int l = 0; l < 3; ++l) { boxOff[l] = off; off += (size_t)8 * Hs[l] * Hs[l] * 36; }

  auto convR = [&](const bf16* inb, const bf16* w_, const float* b_, bf16* o_,
                   int H, int Wd, int Cout) {
    int pix = 8 * H * Wd;
    dim3 g((pix + BM - 1) / BM, (Cout + BN - 1) / BN);
    conv3x3_wmma<true, true><<<g, 256, 0, stream>>>(inb, w_, b_, o_, nullptr, H, Wd, Cout, pix);
  };
  auto convF = [&](const bf16* inb, const bf16* w_, const float* b_, float* o_,
                   int H, int Wd, int Cout) {
    int pix = 8 * H * Wd;
    dim3 g((pix + BM - 1) / BM, (Cout + BN - 1) / BN);
    conv3x3_wmma<false, false><<<g, 256, 0, stream>>>(inb, w_, b_, nullptr, o_, H, Wd, Cout, pix);
  };

  for (int l = 0; l < 3; ++l) {
    const int H = Hs[l], Wd = Hs[l];
    const int nelem = 8 * H * Wd * 256;
    const int cvtg  = (nelem / 4 + 255) / 256;

    // cls branch
    cvt_f32_bf16<<<cvtg, 256, 0, stream>>>(x[l], bufA, nelem);
    convR(bufA, cls_wt[0], cls_b + 0,   bufB, H, Wd, 256);
    convR(bufB, cls_wt[1], cls_b + 256, bufA, H, Wd, 256);
    convR(bufA, cls_wt[2], cls_b + 512, bufB, H, Wd, 256);
    convR(bufB, cls_wt[3], cls_b + 768, bufA, H, Wd, 256);
    convF(bufA, cls_out_wt, cls_out_b, out + clsOff[l], H, Wd, 720);

    // bbox branch (re-convert x, bufA was consumed)
    cvt_f32_bf16<<<cvtg, 256, 0, stream>>>(x[l], bufA, nelem);
    convR(bufA, bbox_wt[0], bbox_b + 0,   bufB, H, Wd, 256);
    convR(bufB, bbox_wt[1], bbox_b + 256, bufA, H, Wd, 256);
    convR(bufA, bbox_wt[2], bbox_b + 512, bufB, H, Wd, 256);
    convR(bufB, bbox_wt[3], bbox_b + 768, bufA, H, Wd, 256);
    convF(bufA, bbox_out_wt, bbox_out_b, out + boxOff[l], H, Wd, 36);
  }

  (void)in_sizes; (void)n_in; (void)out_size; (void)ws_size;
}
